// AEVComputer_52767968199001
// MI455X (gfx1250) — matrix operand order repaired and checked by reading, not compiled, not run on hardware
//
#include <hip/hip_runtime.h>

typedef __attribute__((ext_vector_type(16))) _Float16 v16h;
typedef __attribute__((ext_vector_type(2)))  _Float16 h2v;
typedef __attribute__((ext_vector_type(8)))  float    v8f;

#define NMOL   16
#define NATOM  48
#define NPAIR  1128          // triu(48, k=1)
#define NCHUNK 36            // 36*32 = 1152 (padded pair count)
#define RCR    5.2f
#define RCA    3.5f
#define BIGD   6.2f          // max(RCR,RCA)+1
#define AEVLEN 384
#define PI_F   3.14159265358979323846f

__device__ __forceinline__ float fcutR(float d) {   // cutoff_cosine(d, RCR), div-free
    float f = 0.5f * __cosf(d * (PI_F / RCR)) + 0.5f;
    return (d <= RCR) ? f : 0.0f;
}
__device__ __forceinline__ float fcutA(float d) {   // cutoff_cosine(d, RCA), div-free
    float f = 0.5f * __cosf(d * (PI_F / RCA)) + 0.5f;
    return (d <= RCA) ? f : 0.0f;
}
__device__ __forceinline__ unsigned pkh2(float a, float b) {  // pack 2xf32 -> 2xf16 (RNE)
    h2v h;
    h[0] = (_Float16)a;
    h[1] = (_Float16)b;
    return __builtin_bit_cast(unsigned, h);
}

__global__ __launch_bounds__(32)
void aev_kernel(const int*  __restrict__ species,
                const float* __restrict__ coord,
                const float* __restrict__ pEtaR,
                const float* __restrict__ pShfR,
                const float* __restrict__ pEtaA,
                const float* __restrict__ pZeta,
                const float* __restrict__ pShfA,
                const float* __restrict__ pShfZ,
                float* __restrict__ out)
{
    __shared__ float sx[NATOM], sy[NATOM], sz[NATOM];
    __shared__ int   ssp[NATOM];
    __shared__ float djx[NATOM], djy[NATOM], djz[NATOM], djd[NATOM], fca[NATOM];
    __shared__ unsigned char spv[NATOM];
    __shared__ unsigned char pk[NCHUNK * 32], pl[NCHUNK * 32], psp[NCHUNK * 32];
    __shared__ _Float16 bt[32 * 32];   // staging tile: angular 32 pairs x 32 feats / radial 32 x 16

    const int lane = threadIdx.x;
    const int wg   = blockIdx.x;
    const int m    = wg / NATOM;
    const int j    = wg - m * NATOM;

    // ---- stage molecule coords + species ----
    for (int i = lane; i < NATOM; i += 32) {
        sx[i]  = coord[(m * NATOM + i) * 3 + 0];
        sy[i]  = coord[(m * NATOM + i) * 3 + 1];
        sz[i]  = coord[(m * NATOM + i) * 3 + 2];
        ssp[i] = species[m * NATOM + i];
    }
    __syncthreads();

    // ---- parameters into registers (uniform) ----
    const float etaR = pEtaR[0], etaA = pEtaA[0], zeta = pZeta[0];
    const bool  z32  = (zeta == 32.0f);
    float shfR[16];
#pragma unroll
    for (int f = 0; f < 16; ++f) shfR[f] = pShfR[f];
    float shfA[4];
#pragma unroll
    for (int ia = 0; ia < 4; ++ia) shfA[ia] = pShfA[ia];
    float cZ[8], sZ[8];
#pragma unroll
    for (int iz = 0; iz < 8; ++iz) {
        float s = pShfZ[iz];
        cZ[iz] = __cosf(s);
        sZ[iz] = __sinf(s);
    }

    // ---- per-center neighbor vectors / masked distances / RCA cutoff table ----
    const float xj = sx[j], yj = sy[j], zj = sz[j];
    const bool  vj = (ssp[j] >= 0);
    for (int k = lane; k < NATOM; k += 32) {
        float dx = xj - sx[k], dy = yj - sy[k], dz = zj - sz[k];
        bool  v  = vj && (ssp[k] >= 0) && (k != j);
        float d  = __fsqrt_rn(dx * dx + dy * dy + dz * dz);
        float dm = v ? d : BIGD;
        djx[k] = dx; djy[k] = dy; djz[k] = dz;
        djd[k] = dm;
        fca[k] = fcutA(dm);
        spv[k] = v ? 1 : 0;
    }

    // ---- pair (k,l) LUT + triu species-pair index ----
    for (int p = lane; p < NCHUNK * 32; p += 32) {
        int kk = 0, ll = 0, sidx = 255;
        if (p < NPAIR) {
            int rem = p, k2 = 0;
            while (rem >= NATOM - 1 - k2) { rem -= NATOM - 1 - k2; ++k2; }
            kk = k2; ll = k2 + 1 + rem;
            int a  = min(max(ssp[kk], 0), 3);
            int b  = min(max(ssp[ll], 0), 3);
            int lo = min(a, b), hi = max(a, b);
            sidx = lo * 4 - (lo * (lo - 1)) / 2 + (hi - lo);   // triu_index(S=4)
        }
        pk[p] = (unsigned char)kk;
        pl[p] = (unsigned char)ll;
        psp[p] = (unsigned char)sidx;
    }
    __syncthreads();

    const int row = lane & 15;
    const int kb  = (lane >> 4) * 8;
    const int khi = (lane >> 4) << 4;

    // ================= RADIAL: D[s,f] = sum_k onehot[k,s] * rt[k,f] =================
    v8f accR;
#pragma unroll
    for (int i = 0; i < 8; ++i) accR[i] = 0.0f;

    for (int c = 0; c < 2; ++c) {
        int   kk = c * 32 + lane;
        float d  = (kk < NATOM) ? djd[kk] : BIGD;
        float fc = fcutR(d);
        unsigned upk[8];
#pragma unroll
        for (int f = 0; f < 8; ++f) {
            float t0 = d - shfR[2 * f + 0];
            float t1 = d - shfR[2 * f + 1];
            float v0 = 0.25f * __expf(-etaR * t0 * t0) * fc;
            float v1 = 0.25f * __expf(-etaR * t1 * t1) * fc;
            upk[f] = pkh2(v0, v1);
        }
        {
            uint4* qp = (uint4*)(&bt[lane * 16]);
            qp[0] = make_uint4(upk[0], upk[1], upk[2], upk[3]);
            qp[1] = make_uint4(upk[4], upk[5], upk[6], upk[7]);
        }
        __syncthreads();

        v16h a, b;
#pragma unroll
        for (int e = 0; e < 16; ++e) {          // A: 16x32 f16 layout (row=s, col=K)
            int K  = khi + e;
            int kg = c * 32 + K;
            int sv = (kg < NATOM) ? min(max(ssp[kg], 0), 3) : -1;
            a[e] = (sv == row) ? (_Float16)1.0f : (_Float16)0.0f;
        }
#pragma unroll
        for (int h = 0; h < 16; ++h) {          // B: 32x16 f16 layout (K=h+16*(lane>>4), N=row)
            int K = h + khi;
            b[h] = bt[K * 16 + row];
        }
        accR = __builtin_amdgcn_wmma_f32_16x16x32_f16(false, a, false, b, (short)0, accR, false, false);
        __syncthreads();
    }

    {
        float* obase = out + (size_t)(m * NATOM + j) * AEVLEN;
        int f = row;
#pragma unroll
        for (int v = 0; v < 8; ++v) {
            int s = v + 8 * (lane >> 4);
            if (s < 4) obase[s * 16 + f] = accR[v];
        }
    }

    // ================= ANGULAR: D[s,f] = sum_p onehot[p,s] * at[p,f] =================
    v8f acc0, acc1;
#pragma unroll
    for (int i = 0; i < 8; ++i) { acc0[i] = 0.0f; acc1[i] = 0.0f; }

    for (int c = 0; c < NCHUNK; ++c) {
        int   p  = c * 32 + lane;
        int   kk = pk[p], ll = pl[p];
        bool  tv = (p < NPAIR) && spv[kk] && spv[ll];
        float d1 = tv ? djd[kk] : BIGD;
        float d2 = tv ? djd[ll] : BIGD;
        float dt = tv ? (djx[kk] * djx[ll] + djy[kk] * djy[ll] + djz[kk] * djz[ll]) : 0.0f;

        float ca   = 0.95f * dt * __builtin_amdgcn_rcpf(fmaxf(d1, 1e-8f) * fmaxf(d2, 1e-8f));
        float sa   = __fsqrt_rn(fmaxf(1.0f - ca * ca, 0.0f));   // sin(arccos(ca)) >= 0
        float pref = tv ? (2.0f * fca[kk] * fca[ll]) : 0.0f;
        float dm   = 0.5f * (d1 + d2);

        unsigned upk[16];
        if (z32) {                                  // uniform branch: zeta == 32 -> 5 squarings
#pragma unroll
            for (int ia = 0; ia < 4; ++ia) {
                float t2 = dm - shfA[ia];
                float f2 = __expf(-etaA * t2 * t2) * pref;
#pragma unroll
                for (int iz = 0; iz < 8; iz += 2) {
                    float cv0 = ca * cZ[iz]     + sa * sZ[iz];
                    float cv1 = ca * cZ[iz + 1] + sa * sZ[iz + 1];
                    float u0 = 0.5f * (1.0f + cv0), u1 = 0.5f * (1.0f + cv1);
                    u0 *= u0; u0 *= u0; u0 *= u0; u0 *= u0; u0 *= u0;   // ^32
                    u1 *= u1; u1 *= u1; u1 *= u1; u1 *= u1; u1 *= u1;
                    upk[ia * 4 + (iz >> 1)] = pkh2(u0 * f2, u1 * f2);
                }
            }
        } else {
#pragma unroll
            for (int ia = 0; ia < 4; ++ia) {
                float t2 = dm - shfA[ia];
                float f2 = __expf(-etaA * t2 * t2) * pref;
#pragma unroll
                for (int iz = 0; iz < 8; iz += 2) {
                    float cv0 = ca * cZ[iz]     + sa * sZ[iz];
                    float cv1 = ca * cZ[iz + 1] + sa * sZ[iz + 1];
                    float u0 = __powf(0.5f * (1.0f + cv0), zeta);
                    float u1 = __powf(0.5f * (1.0f + cv1), zeta);
                    upk[ia * 4 + (iz >> 1)] = pkh2(u0 * f2, u1 * f2);
                }
            }
        }
        {
            uint4* qp = (uint4*)(&bt[lane * 32]);
            qp[0] = make_uint4(upk[0],  upk[1],  upk[2],  upk[3]);
            qp[1] = make_uint4(upk[4],  upk[5],  upk[6],  upk[7]);
            qp[2] = make_uint4(upk[8],  upk[9],  upk[10], upk[11]);
            qp[3] = make_uint4(upk[12], upk[13], upk[14], upk[15]);
        }
        __syncthreads();

        v16h a, b0, b1;
#pragma unroll
        for (int e = 0; e < 16; ++e) {          // A: one-hot of species-pair index
            int K  = khi + e;
            int sp = psp[c * 32 + K];
            a[e] = (sp == row) ? (_Float16)1.0f : (_Float16)0.0f;
        }
#pragma unroll
        for (int h = 0; h < 16; ++h) {          // B tiles: feats 0-15 and 16-31
            int K = h + khi;
            b0[h] = bt[K * 32 + row];
            b1[h] = bt[K * 32 + 16 + row];
        }
        acc0 = __builtin_amdgcn_wmma_f32_16x16x32_f16(false, a, false, b0, (short)0, acc0, false, false);
        acc1 = __builtin_amdgcn_wmma_f32_16x16x32_f16(false, a, false, b1, (short)0, acc1, false, false);
        __syncthreads();
    }

    {
        float* obase = out + (size_t)(m * NATOM + j) * AEVLEN + 64;
        int f = row;
#pragma unroll
        for (int v = 0; v < 8; ++v) {
            int s = v + 8 * (lane >> 4);
            if (s < 10) {
                obase[s * 32 + f]      = acc0[v];
                obase[s * 32 + 16 + f] = acc1[v];
            }
        }
    }
}

extern "C" void kernel_launch(void* const* d_in, const int* in_sizes, int n_in,
                              void* d_out, int out_size, void* d_ws, size_t ws_size,
                              hipStream_t stream) {
    (void)in_sizes; (void)n_in; (void)d_ws; (void)ws_size; (void)out_size;
    const int*   species = (const int*)d_in[0];
    const float* coord   = (const float*)d_in[1];
    const float* etaR    = (const float*)d_in[2];
    const float* shfR    = (const float*)d_in[3];
    const float* etaA    = (const float*)d_in[4];
    const float* zeta    = (const float*)d_in[5];
    const float* shfA    = (const float*)d_in[6];
    const float* shfZ    = (const float*)d_in[7];

    aev_kernel<<<NMOL * NATOM, 32, 0, stream>>>(species, coord, etaR, shfR,
                                                etaA, zeta, shfA, shfZ,
                                                (float*)d_out);
}
